// ProtoSSMv2_18846316495016
// MI455X (gfx1250) — compile-verified, matrix operand order checked by reading, not run on hardware
//
#include <hip/hip_runtime.h>
#include <math.h>

#define D_IN  1536
#define DM    256
#define NST   16
#define LL    2
#define NCLS  234
#define NHD   4
#define DH    64
#define DCV   4
#define BSZ   256
#define TLEN  64
#define MROWS (BSZ*TLEN)   // 16384

typedef __attribute__((ext_vector_type(16))) _Float16 v16h;
typedef __attribute__((ext_vector_type(8)))  _Float16 v8h;
typedef __attribute__((ext_vector_type(8)))  float    v8f;

__device__ __forceinline__ float gelu_f(float x) {
    return 0.5f * x * (1.0f + erff(x * 0.70710678118654752f));
}
__device__ __forceinline__ float softplus_f(float x) {
    return (x > 20.0f) ? x : log1pf(__expf(x));
}
__device__ __forceinline__ float sigmoid_f(float x) {
    return 1.0f / (1.0f + __expf(-x));
}

// Load one 16x32 f16 fragment slice for this lane (base uniform in SGPRs,
// 32-bit lane offset -> GVS-form global_load_b128 pairs).
// Per ISA layout: lane holds K = kb..kb+7 (lo) and kb+16..kb+23 (hi),
// kb = (lane>>4)*8.
__device__ __forceinline__ v16h load_frag(const _Float16* __restrict__ base, int off) {
    union { v16h v; v8h h[2]; } u;
    u.h[0] = *(const v8h*)(base + off);
    u.h[1] = *(const v8h*)(base + off + 16);
    return u.v;
}

// ---------------------------------------------------------------------------
// WMMA GEMM, f16 operands, f32 accumulate. Each wave computes a (MT*16)x(NT*16)
// block: MT A-fragments and NT B-fragments feed MT*NT wmmas per K=32 step.
// out (f32) and/or outh (f16) may be null. ACT (compile-time): 0=none, 1=gelu,
// 2=softplus, 3=proto (acc*softplus(*scl)+bias). M%(16*MT)==0, K%32==0.
// ---------------------------------------------------------------------------
template<int MT, int NT, int ACT>
__global__ void __launch_bounds__(256)
k_wmma_gemm(const _Float16* __restrict__ A, const _Float16* __restrict__ W,
            const float* __restrict__ bias, const float* __restrict__ res,
            const float* __restrict__ scl,
            float* __restrict__ out, _Float16* __restrict__ outh,
            int M, int N, int K)
{
    int lane = threadIdx.x & 31;
    int wid  = blockIdx.x * (blockDim.x >> 5) + (threadIdx.x >> 5);
    int Mtiles = M >> 4;
    int Ntiles = (N + 15) >> 4;
    int Mg = Mtiles / MT;
    int Ng = (Ntiles + NT - 1) / NT;
    if (wid >= Mg * Ng) return;
    int mg = wid / Ng, ng = wid % Ng;
    int kb = (lane >> 4) << 3;

    int aoff[MT];
    #pragma unroll
    for (int i = 0; i < MT; ++i) {
        int mA = ((mg * MT + i) << 4) + (lane & 15);
        aoff[i] = mA * K + kb;
    }
    int woff[NT];
    #pragma unroll
    for (int j = 0; j < NT; ++j) {
        int nW = ((ng * NT + j) << 4) + (lane & 15);
        if (nW > N - 1) nW = N - 1;        // clamp ragged N (stores guarded)
        woff[j] = nW * K + kb;
    }

    v8f c[MT][NT];
    #pragma unroll
    for (int i = 0; i < MT; ++i)
        #pragma unroll
        for (int j = 0; j < NT; ++j)
            c[i][j] = (v8f){0.f,0.f,0.f,0.f,0.f,0.f,0.f,0.f};

    for (int k0 = 0; k0 < K; k0 += 32) {
        v16h a[MT], b[NT];
        #pragma unroll
        for (int i = 0; i < MT; ++i) a[i] = load_frag(A, aoff[i] + k0);
        #pragma unroll
        for (int j = 0; j < NT; ++j) b[j] = load_frag(W, woff[j] + k0);
        #pragma unroll
        for (int i = 0; i < MT; ++i)
            #pragma unroll
            for (int j = 0; j < NT; ++j)
                c[i][j] = __builtin_amdgcn_wmma_f32_16x16x32_f16(
                    false, a[i], false, b[j], (short)0, c[i][j], false, false);
    }

    float sp = 1.0f;
    if (ACT == 3) sp = softplus_f(scl[0]);
    #pragma unroll
    for (int i = 0; i < MT; ++i) {
        int mbase = ((mg * MT + i) << 4) + ((lane >> 4) << 3);
        #pragma unroll
        for (int j = 0; j < NT; ++j) {
            int ncol = ((ng * NT + j) << 4) + (lane & 15);
            int nbi = (ncol < N) ? ncol : (N - 1);
            #pragma unroll
            for (int r = 0; r < 8; ++r) {
                int m = mbase + r;
                float v = c[i][j][r];
                if (ACT == 3) v *= sp;
                if (bias) v += bias[nbi];
                if (ACT == 1) v = gelu_f(v);
                else if (ACT == 2) v = softplus_f(v);
                if (res) v += res[(size_t)m * N + ((ncol < N) ? ncol : 0)];
                if (ncol < N) {
                    if (out)  out[(size_t)m * N + ncol] = v;
                    if (outh) outh[(size_t)m * N + ncol] = (_Float16)v;
                }
            }
        }
    }
}

// ---------------------------------------------------------------------------
// Row-wise (D=256, one block per row) reductions
// ---------------------------------------------------------------------------
__device__ __forceinline__ float block_sum(float v, float* red) {
    int tid = threadIdx.x;
    red[tid] = v; __syncthreads();
    for (int s = 128; s > 0; s >>= 1) {
        if (tid < s) red[tid] += red[tid + s];
        __syncthreads();
    }
    float r = red[0]; __syncthreads();
    return r;
}

// h = gelu(LN(y)) + pos[t] + meta[b]; writes f32 + f16
__global__ void k_ln_gelu_pm(const float* __restrict__ y, const float* __restrict__ g,
                             const float* __restrict__ bv, const float* __restrict__ pos,
                             const float* __restrict__ meta,
                             float* __restrict__ outf, _Float16* __restrict__ outh)
{
    __shared__ float red[256];
    int row = blockIdx.x, tid = threadIdx.x;
    float v = y[(size_t)row * DM + tid];
    float mean = block_sum(v, red) * (1.0f / DM);
    float dv = v - mean;
    float var = block_sum(dv * dv, red) * (1.0f / DM);
    float xn = dv * rsqrtf(var + 1e-5f) * g[tid] + bv[tid];
    int t = row & (TLEN - 1), b = row >> 6;
    float o = gelu_f(xn) + pos[t * DM + tid] + meta[b * DM + tid];
    outf[(size_t)row * DM + tid] = o;
    outh[(size_t)row * DM + tid] = (_Float16)o;
}

// h = LN(x + res); writes f32 + f16
__global__ void k_add_ln(const float* __restrict__ x, const float* __restrict__ r0,
                         const float* __restrict__ g, const float* __restrict__ bv,
                         float* __restrict__ outf, _Float16* __restrict__ outh)
{
    __shared__ float red[256];
    int row = blockIdx.x, tid = threadIdx.x;
    float v = x[(size_t)row * DM + tid] + r0[(size_t)row * DM + tid];
    float mean = block_sum(v, red) * (1.0f / DM);
    float dv = v - mean;
    float var = block_sum(dv * dv, red) * (1.0f / DM);
    float o = dv * rsqrtf(var + 1e-5f) * g[tid] + bv[tid];
    outf[(size_t)row * DM + tid] = o;
    outh[(size_t)row * DM + tid] = (_Float16)o;
}

// LN -> f16 only (feeds a GEMM)
__global__ void k_ln(const float* __restrict__ x, const float* __restrict__ g,
                     const float* __restrict__ bv, _Float16* __restrict__ outh)
{
    __shared__ float red[256];
    int row = blockIdx.x, tid = threadIdx.x;
    float v = x[(size_t)row * DM + tid];
    float mean = block_sum(v, red) * (1.0f / DM);
    float dv = v - mean;
    float var = block_sum(dv * dv, red) * (1.0f / DM);
    outh[(size_t)row * DM + tid] = (_Float16)(dv * rsqrtf(var + 1e-5f) * g[tid] + bv[tid]);
}

// L2 normalize rows -> f16 (feeds a GEMM as A or W)
__global__ void k_l2norm(const float* __restrict__ x, _Float16* __restrict__ outh)
{
    __shared__ float red[256];
    int row = blockIdx.x, tid = threadIdx.x;
    float v = x[(size_t)row * DM + tid];
    float ss = block_sum(v * v, red);
    float nrm = fmaxf(sqrtf(ss), 1e-12f);
    outh[(size_t)row * DM + tid] = (_Float16)(v / nrm);
}

// ---------------------------------------------------------------------------
// Elementwise / small kernels
// ---------------------------------------------------------------------------
__global__ void k_cvt(const float* __restrict__ in, _Float16* __restrict__ out, int n)
{
    int i = blockIdx.x * blockDim.x + threadIdx.x;
    if (i < n) out[i] = (_Float16)in[i];
}

__global__ void k_meta_gather(const int* __restrict__ site_ids, const int* __restrict__ hours,
                              const float* __restrict__ site_emb, const float* __restrict__ hour_emb,
                              _Float16* __restrict__ out)
{
    int i = blockIdx.x * blockDim.x + threadIdx.x;
    if (i >= BSZ * 32) return;
    int b = i >> 5, j = i & 31;
    float v = (j < 16) ? site_emb[site_ids[b] * 16 + j]
                       : hour_emb[hours[b] * 16 + (j - 16)];
    out[i] = (_Float16)v;
}

// time-reverse h -> f32 (scan input) and f16 (GEMM input)
__global__ void k_flip(const float* __restrict__ in, float* __restrict__ outf,
                       _Float16* __restrict__ outh)
{
    int i = blockIdx.x * blockDim.x + threadIdx.x;
    if (i >= MROWS * DM) return;
    int d = i & (DM - 1);
    int bt = i >> 8;
    int t = bt & (TLEN - 1), b = bt >> 6;
    size_t o = (size_t)((b << 6) + (TLEN - 1 - t)) * DM + d;
    float v = in[i];
    outf[o] = v;
    outh[o] = (_Float16)v;
}

// depthwise causal conv (DC=4, left pad) + SiLU -> f16 (feeds 3 GEMMs)
__global__ void k_conv_silu(const float* __restrict__ x, const float* __restrict__ w,
                            const float* __restrict__ cb, _Float16* __restrict__ out)
{
    int i = blockIdx.x * blockDim.x + threadIdx.x;
    if (i >= MROWS * DM) return;
    int d = i & (DM - 1);
    int bt = i >> 8;
    int t = bt & (TLEN - 1);
    float acc = cb[d];
    #pragma unroll
    for (int k = 0; k < DCV; ++k) {
        int tt = t + k - (DCV - 1);
        if (tt >= 0) acc += w[d * DCV + k] * x[(size_t)(bt + k - (DCV - 1)) * DM + d];
    }
    out[i] = (_Float16)(acc * sigmoid_f(acc));
}

// Selective scan: one thread per (b,d); N=16 state in VGPRs. Writes the f16
// concat buffer (ldo=2D) at column `off`; rev flips time. Folds "+ x*Dp".
__global__ void k_scan(const float* __restrict__ xin, const float* __restrict__ dt,
                       const float* __restrict__ Bm, const float* __restrict__ Cm,
                       const float* __restrict__ A_log, const float* __restrict__ Dp,
                       _Float16* __restrict__ out, int ldo, int off, int rev)
{
    int i = blockIdx.x * blockDim.x + threadIdx.x;
    if (i >= BSZ * DM) return;
    int d = i & (DM - 1);
    int b = i >> 8;
    float An[NST], hs[NST];
    #pragma unroll
    for (int n = 0; n < NST; ++n) { An[n] = -__expf(A_log[d * NST + n]); hs[n] = 0.f; }
    float dp = Dp[d];
    for (int t = 0; t < TLEN; ++t) {
        size_t row = (size_t)(b * TLEN + t);
        float dtv = dt[row * DM + d];
        float xv  = xin[row * DM + d];
        float xdt = xv * dtv;
        float y = 0.f;
        #pragma unroll
        for (int n = 0; n < NST; ++n) {
            float dA = __expf(An[n] * dtv);
            hs[n] = hs[n] * dA + xdt * Bm[row * NST + n];
            y += hs[n] * Cm[row * NST + n];
        }
        int tt = rev ? (TLEN - 1 - t) : t;
        out[(size_t)(b * TLEN + tt) * ldo + off + d] = (_Float16)(y + xv * dp);
    }
}

// Attention: one block per (batch, head). T=64, dh=64; scores/softmax in LDS.
__global__ void k_attn(const float* __restrict__ qkv, _Float16* __restrict__ out)
{
    __shared__ float qs[TLEN][DH + 1];   // Q, later reused for V
    __shared__ float ks[TLEN][DH + 1];
    __shared__ float ss[TLEN][TLEN + 1];
    int b = blockIdx.x >> 2;
    int hh = blockIdx.x & 3;
    int tid = threadIdx.x;
    #pragma unroll
    for (int it = 0; it < 16; ++it) {
        int idx = tid + (it << 8);
        int r = idx >> 6, cc = idx & 63;
        size_t base = (size_t)(b * TLEN + r) * (3 * DM) + hh * DH + cc;
        qs[r][cc] = qkv[base];
        ks[r][cc] = qkv[base + DM];
    }
    __syncthreads();
    #pragma unroll
    for (int it = 0; it < 16; ++it) {
        int idx = tid + (it << 8);
        int r = idx >> 6, cc = idx & 63;
        float acc = 0.f;
        for (int j = 0; j < DH; ++j) acc += qs[r][j] * ks[cc][j];
        ss[r][cc] = acc * 0.125f;   // 1/sqrt(64)
    }
    __syncthreads();
    if (tid < TLEN) {
        float mx = -1e30f;
        for (int cc = 0; cc < TLEN; ++cc) mx = fmaxf(mx, ss[tid][cc]);
        float sum = 0.f;
        for (int cc = 0; cc < TLEN; ++cc) { float e = __expf(ss[tid][cc] - mx); ss[tid][cc] = e; sum += e; }
        float inv = 1.0f / sum;
        for (int cc = 0; cc < TLEN; ++cc) ss[tid][cc] *= inv;
    }
    __syncthreads();
    #pragma unroll
    for (int it = 0; it < 16; ++it) {     // reload V into qs
        int idx = tid + (it << 8);
        int r = idx >> 6, cc = idx & 63;
        qs[r][cc] = qkv[(size_t)(b * TLEN + r) * (3 * DM) + 2 * DM + hh * DH + cc];
    }
    __syncthreads();
    #pragma unroll
    for (int it = 0; it < 16; ++it) {
        int idx = tid + (it << 8);
        int r = idx >> 6, cc = idx & 63;
        float acc = 0.f;
        for (int j = 0; j < TLEN; ++j) acc += ss[r][j] * qs[j][cc];
        out[(size_t)(b * TLEN + r) * DM + hh * DH + cc] = (_Float16)acc;
    }
}

// logits = sigmoid(fa[c]) * sim + (1 - sigmoid(fa[c])) * perch (in place)
__global__ void k_fuse(float* __restrict__ sim, const float* __restrict__ perch,
                       const float* __restrict__ fa)
{
    int i = blockIdx.x * blockDim.x + threadIdx.x;
    if (i >= MROWS * NCLS) return;
    int c = i % NCLS;
    float a = sigmoid_f(fa[c]);
    sim[i] = a * sim[i] + (1.0f - a) * perch[i];
}

// ---------------------------------------------------------------------------
extern "C" void kernel_launch(void* const* d_in, const int* in_sizes, int n_in,
                              void* d_out, int out_size, void* d_ws, size_t ws_size,
                              hipStream_t stream)
{
    (void)in_sizes; (void)n_in; (void)out_size; (void)ws_size;
    const float* emb         = (const float*)d_in[0];
    const float* perch       = (const float*)d_in[1];
    const int*   site_ids    = (const int*)  d_in[2];
    const int*   hours       = (const int*)  d_in[3];
    const float* in_w        = (const float*)d_in[4];
    const float* in_b        = (const float*)d_in[5];
    const float* ln0_g       = (const float*)d_in[6];
    const float* ln0_b       = (const float*)d_in[7];
    const float* pos_enc     = (const float*)d_in[8];
    const float* site_emb    = (const float*)d_in[9];
    const float* hour_emb    = (const float*)d_in[10];
    const float* meta_w      = (const float*)d_in[11];
    const float* meta_b      = (const float*)d_in[12];
    const float* ssm_in_w    = (const float*)d_in[13];
    const float* ssm_conv_w  = (const float*)d_in[14];
    const float* ssm_conv_b  = (const float*)d_in[15];
    const float* ssm_dt_w    = (const float*)d_in[16];
    const float* ssm_dt_b    = (const float*)d_in[17];
    const float* ssm_A_log   = (const float*)d_in[18];
    const float* ssm_Dp      = (const float*)d_in[19];
    const float* ssm_B_w     = (const float*)d_in[20];
    const float* ssm_C_w     = (const float*)d_in[21];
    const float* merge_w     = (const float*)d_in[22];
    const float* merge_b     = (const float*)d_in[23];
    const float* ln_g        = (const float*)d_in[24];
    const float* ln_b        = (const float*)d_in[25];
    const float* qkv_w       = (const float*)d_in[26];
    const float* qkv_b       = (const float*)d_in[27];
    const float* ao_w        = (const float*)d_in[28];
    const float* ao_b        = (const float*)d_in[29];
    const float* ca_ln1_g    = (const float*)d_in[30];
    const float* ca_ln1_b    = (const float*)d_in[31];
    const float* ffn_w1      = (const float*)d_in[32];
    const float* ffn_b1      = (const float*)d_in[33];
    const float* ffn_w2      = (const float*)d_in[34];
    const float* ffn_b2      = (const float*)d_in[35];
    const float* ca_ln2_g    = (const float*)d_in[36];
    const float* ca_ln2_b    = (const float*)d_in[37];
    const float* prototypes  = (const float*)d_in[38];
    const float* proto_temp  = (const float*)d_in[39];
    const float* class_bias  = (const float*)d_in[40];
    const float* fusion_a    = (const float*)d_in[41];

    const size_t MD = (size_t)MROWS * DM;

    // ---- f32 scratch -------------------------------------------------------
    float* ws = (float*)d_ws;
    float* h    = ws;                    // [M,D]   residual stream
    float* t0   = h   + MD;              // [M,D]   GEMM f32 out (pre-LN)
    float* t1   = t0  + MD;              // [M,D]   reversed h (scan input)
    float* t2   = t1  + MD;              // [M,D]   x_ssm (conv input)
    float* t4   = t2  + MD;              // [M,D]   dt
    float* big  = t4  + MD;              // [M,3D]  qkv f32
    float* Bmb  = big + 3 * MD;          // [M,16]
    float* Cmb  = Bmb + (size_t)MROWS * NST;
    float* metaf = Cmb + (size_t)MROWS * NST;    // [B,D]
    float* f32end = metaf + (size_t)BSZ * DM;

    // ---- f16 scratch -------------------------------------------------------
    _Float16* fh = (_Float16*)f32end;
    _Float16* embh  = fh;                          // [M,1536]
    _Float16* hh    = embh + (size_t)MROWS * D_IN; // [M,D]
    _Float16* t1h   = hh   + MD;                   // [M,D] flipped h / attn out
    _Float16* t3h   = t1h  + MD;                   // [M,D] conv+silu out
    _Float16* t0h   = t3h  + MD;                   // [M,D] LN / l2norm out
    _Float16* conch = t0h  + MD;                   // [M,2D] scan concat
    _Float16* bigh  = conch + 2 * MD;              // [M,2D] ffn1 out
    _Float16* metainh = bigh + 2 * MD;             // [B,32]
    _Float16* protnh  = metainh + BSZ * 32;        // [234,D]
    _Float16* w16 = protnh + (size_t)NCLS * DM;
    _Float16* w_in   = w16;                         w16 += (size_t)DM * D_IN;
    _Float16* w_meta = w16;                         w16 += (size_t)DM * 32;
    _Float16* w_sin  = w16;                         w16 += (size_t)LL * 2 * 2 * DM * DM;
    _Float16* w_dt   = w16;                         w16 += (size_t)LL * 2 * DM * DM;
    _Float16* w_B    = w16;                         w16 += (size_t)LL * 2 * NST * DM;
    _Float16* w_C    = w16;                         w16 += (size_t)LL * 2 * NST * DM;
    _Float16* w_mrg  = w16;                         w16 += (size_t)LL * DM * 2 * DM;
    _Float16* w_qkv  = w16;                         w16 += (size_t)3 * DM * DM;
    _Float16* w_ao   = w16;                         w16 += (size_t)DM * DM;
    _Float16* w_f1   = w16;                         w16 += (size_t)2 * DM * DM;
    _Float16* w_f2   = w16;                         w16 += (size_t)2 * DM * DM;

    const int EW = 256;
    auto nb = [](size_t n, int b) { return (int)((n + b - 1) / b); };
    auto cvt = [&](const float* s, _Float16* dst, size_t n) {
        k_cvt<<<nb(n, EW), EW, 0, stream>>>(s, dst, (int)n);
    };
    auto gemm24 = [&](const _Float16* A, const _Float16* W, const float* bias,
                      const float* res, const float* scl, float* out, _Float16* outh,
                      int M, int N, int K, int act) {
        int Mg = (M >> 4) / 2, Ng = (((N + 15) >> 4) + 3) / 4;
        int blocks = (Mg * Ng + 7) / 8;
        switch (act) {
        case 1:
            k_wmma_gemm<2, 4, 1><<<blocks, 256, 0, stream>>>(A, W, bias, res, scl, out, outh, M, N, K);
            break;
        case 2:
            k_wmma_gemm<2, 4, 2><<<blocks, 256, 0, stream>>>(A, W, bias, res, scl, out, outh, M, N, K);
            break;
        case 3:
            k_wmma_gemm<2, 4, 3><<<blocks, 256, 0, stream>>>(A, W, bias, res, scl, out, outh, M, N, K);
            break;
        default:
            k_wmma_gemm<2, 4, 0><<<blocks, 256, 0, stream>>>(A, W, bias, res, scl, out, outh, M, N, K);
            break;
        }
    };
    auto gemm21 = [&](const _Float16* A, const _Float16* W, float* out,
                      int M, int N, int K) {
        int Mg = (M >> 4) / 2, Ng = (N + 15) >> 4;
        k_wmma_gemm<2, 1, 0><<<(Mg * Ng + 7) / 8, 256, 0, stream>>>(
            A, W, nullptr, nullptr, nullptr, out, nullptr, M, N, K);
    };

    // ---- one-time f16 conversions -----------------------------------------
    cvt(emb, embh, (size_t)MROWS * D_IN);
    cvt(in_w, w_in, (size_t)DM * D_IN);
    cvt(meta_w, w_meta, (size_t)DM * 32);
    cvt(ssm_in_w, w_sin, (size_t)LL * 2 * 2 * DM * DM);
    cvt(ssm_dt_w, w_dt, (size_t)LL * 2 * DM * DM);
    cvt(ssm_B_w, w_B, (size_t)LL * 2 * NST * DM);
    cvt(ssm_C_w, w_C, (size_t)LL * 2 * NST * DM);
    cvt(merge_w, w_mrg, (size_t)LL * DM * 2 * DM);
    cvt(qkv_w, w_qkv, (size_t)3 * DM * DM);
    cvt(ao_w, w_ao, (size_t)DM * DM);
    cvt(ffn_w1, w_f1, (size_t)2 * DM * DM);
    cvt(ffn_w2, w_f2, (size_t)2 * DM * DM);

    // ---- input projection + LN + GELU + pos + meta ------------------------
    gemm24(embh, w_in, in_b, nullptr, nullptr, t0, nullptr, MROWS, DM, D_IN, 0);
    k_meta_gather<<<nb(BSZ * 32, EW), EW, 0, stream>>>(site_ids, hours, site_emb, hour_emb, metainh);
    gemm24(metainh, w_meta, meta_b, nullptr, nullptr, metaf, nullptr, BSZ, DM, 32, 0);
    k_ln_gelu_pm<<<MROWS, EW, 0, stream>>>(t0, ln0_g, ln0_b, pos_enc, metaf, h, hh);

    // ---- bidirectional SSM layers -----------------------------------------
    for (int l = 0; l < LL; ++l) {
        for (int dir = 0; dir < 2; ++dir) {
            const float* xin = h;
            const _Float16* xinh = hh;
            if (dir == 1) {
                k_flip<<<nb(MD, EW), EW, 0, stream>>>(h, t1, t1h);
                xin = t1; xinh = t1h;
            }
            size_t idx = (size_t)(l * 2 + dir);
            gemm24(xinh, w_sin + idx * 2 * DM * DM, nullptr, nullptr, nullptr,
                   t2, nullptr, MROWS, DM, DM, 0);                       // x_ssm
            k_conv_silu<<<nb(MD, EW), EW, 0, stream>>>(
                   t2, ssm_conv_w + idx * DM * DCV, ssm_conv_b + idx * DM, t3h);
            gemm24(t3h, w_dt + idx * DM * DM, ssm_dt_b + idx * DM, nullptr, nullptr,
                   t4, nullptr, MROWS, DM, DM, 2);                       // softplus(dt)
            gemm21(t3h, w_B + idx * NST * DM, Bmb, MROWS, NST, DM);
            gemm21(t3h, w_C + idx * NST * DM, Cmb, MROWS, NST, DM);
            k_scan<<<nb((size_t)BSZ * DM, EW), EW, 0, stream>>>(
                   xin, t4, Bmb, Cmb, ssm_A_log + idx * DM * NST, ssm_Dp + idx * DM,
                   conch, 2 * DM, dir * DM, dir);                        // -> concat half
        }
        gemm24(conch, w_mrg + (size_t)l * DM * 2 * DM, merge_b + (size_t)l * DM,
               nullptr, nullptr, t0, nullptr, MROWS, DM, 2 * DM, 0);
        k_add_ln<<<MROWS, EW, 0, stream>>>(t0, h, ln_g + (size_t)l * DM,
                                           ln_b + (size_t)l * DM, h, hh);
    }

    // ---- temporal cross-attention -----------------------------------------
    k_ln<<<MROWS, EW, 0, stream>>>(h, ca_ln1_g, ca_ln1_b, t0h);
    gemm24(t0h, w_qkv, qkv_b, nullptr, nullptr, big, nullptr, MROWS, 3 * DM, DM, 0);
    k_attn<<<BSZ * NHD, 256, 0, stream>>>(big, t1h);
    gemm24(t1h, w_ao, ao_b, h, nullptr, h, nullptr, MROWS, DM, DM, 0);   // h += proj

    k_ln<<<MROWS, EW, 0, stream>>>(h, ca_ln2_g, ca_ln2_b, t0h);
    gemm24(t0h, w_f1, ffn_b1, nullptr, nullptr, nullptr, bigh, MROWS, 2 * DM, DM, 1);
    gemm24(bigh, w_f2, ffn_b2, h, nullptr, h, nullptr, MROWS, DM, 2 * DM, 0);

    // ---- prototype head ----------------------------------------------------
    k_l2norm<<<MROWS, EW, 0, stream>>>(h, t0h);
    k_l2norm<<<NCLS, EW, 0, stream>>>(prototypes, protnh);
    float* simout = (float*)d_out;
    gemm24(t0h, protnh, class_bias, nullptr, proto_temp, simout, nullptr,
           MROWS, NCLS, DM, 3);
    k_fuse<<<nb((size_t)MROWS * NCLS, EW), EW, 0, stream>>>(simout, perch, fusion_a);

    // second output: h
    hipMemcpyAsync(simout + (size_t)MROWS * NCLS, h, MD * sizeof(float),
                   hipMemcpyDeviceToDevice, stream);
}
